// Block_62362925138693
// MI455X (gfx1250) — matrix operand order repaired and checked by reading, not compiled, not run on hardware
//
#include <hip/hip_runtime.h>

// Transformer block for MI455X (gfx1250, wave32, WMMA).
// - All GEMMs: v_wmma_f32_16x16x32_bf16 (bf16 in, f32 accumulate).
// - Weight GEMMs (QKV / FF1 / FF2): 4-wave workgroups compute a 64x64 output
//   tile; the shared 64x32 weight chunk is staged into LDS with gfx1250
//   GLOBAL_LOAD_ASYNC_TO_LDS_B128 (ASYNCcnt-tracked, double-buffered) and
//   consumed with contiguous ds_load_b128 B-fragments. A operands are
//   software-pipelined in registers one chunk ahead; the k-loop is unrolled
//   2x so the ping-pong buffers use compile-time LDS offsets.
// - Weights are transposed during the one-time fp32->bf16 conversion and V is
//   produced transposed so every WMMA operand streams 16B/32B per lane.

#define B_  2
#define T_  2048
#define D_  1024
#define H_  16
#define HS_ 64
#define NR  (B_ * T_)    // 4096 rows
#define DFF (4 * D_)     // 4096

#define BROWP 40                       // padded LDS row pitch (elems) = 80B
#define BBUFE (64 * BROWP)             // one B-tile buffer, elements
#define BBUFB (BBUFE * 2)              // one B-tile buffer, bytes

typedef __bf16 bf16_t;
typedef __attribute__((ext_vector_type(16))) __bf16 v16bf;
typedef __attribute__((ext_vector_type(8)))  __bf16 v8bf;
typedef __attribute__((ext_vector_type(8)))  float  v8f;

__device__ __forceinline__ v8f wmma_bf16(v16bf a, v16bf b, v8f c) {
  return __builtin_amdgcn_wmma_f32_16x16x32_bf16(false, a, false, b, (short)0, c,
                                                 false, false);
}

// Async 16B global->LDS copy (ASYNCcnt tracked). lds_off is the byte offset of
// the destination inside the workgroup's LDS allocation (our single __shared__
// array sits at offset 0). lds_escape makes the buffer's address escape so the
// compiler must assume the asm writes it.
__device__ __forceinline__ void async_g2l_b128(unsigned lds_off, const void* gaddr,
                                               const void* lds_escape) {
  asm volatile("global_load_async_to_lds_b128 %0, %1, off"
               :: "v"(lds_off), "v"(gaddr), "v"(lds_escape) : "memory");
}
__device__ __forceinline__ void wait_async_le2() {
  asm volatile("s_wait_asynccnt 0x2" ::: "memory");
}
__device__ __forceinline__ void wait_async_0() {
  asm volatile("s_wait_asynccnt 0x0" ::: "memory");
}

// ---- WMMA fragment loaders (contiguous k = 16*(lane/16) + s in A and B) ------------------
// A fragment: 16x32 bf16 tile from row-major source; one 32B run per lane.
__device__ __forceinline__ v16bf frag_a(const bf16_t* __restrict__ base, int ld, int lane) {
  const int row = lane & 15;
  const int kb  = (lane >> 4) << 4;           // 0 or 16
  const bf16_t* p = base + (size_t)row * ld + kb;
  v16bf a;
#pragma unroll
  for (int i = 0; i < 16; ++i) a[i] = p[i];
  return a;
}

// B fragment from the LDS-staged transposed weight tile [64][BROWP].
__device__ __forceinline__ v16bf frag_bT_lds(const bf16_t* base, int lane) {
  const int col = lane & 15;
  const int kh  = (lane >> 4) << 4;           // 0 or 16
  const bf16_t* p = base + col * BROWP + kh;
  v16bf b;
#pragma unroll
  for (int i = 0; i < 16; ++i) b[i] = p[i];
  return b;
}

// B fragment for Q*K^T: contraction = head-dim e (contiguous); N = key.
__device__ __forceinline__ v16bf frag_b_kT(const bf16_t* __restrict__ kbase,
                                           int key0, int e0, int lane) {
  const int key = key0 + (lane & 15);
  const int e   = e0 + ((lane >> 4) << 4);
  const bf16_t* p = kbase + (size_t)key * HS_ + e;
  v16bf b;
#pragma unroll
  for (int i = 0; i < 16; ++i) b[i] = p[i];
  return b;
}

// B fragment for P*V from transposed V (vT[e][t]); K = key contiguous.
__device__ __forceinline__ v16bf frag_b_vT(const bf16_t* __restrict__ vTb,
                                           int key0, int e0, int lane) {
  const int e  = e0 + (lane & 15);
  const int kh = key0 + ((lane >> 4) << 4);
  const bf16_t* p = vTb + (size_t)e * T_ + kh;
  v16bf b;
#pragma unroll
  for (int i = 0; i < 16; ++i) b[i] = p[i];
  return b;
}

// Stage one 64x32 bf16 chunk of a transposed weight matrix WT[N][K] into LDS.
// 256 16B segments, 2 per thread (128 threads).
__device__ __forceinline__ void stage_btile(const bf16_t* __restrict__ wT, int ldk,
                                            int n0, int kk, int buf, int tid,
                                            const bf16_t* lds_base) {
#pragma unroll
  for (int j = 0; j < 2; ++j) {
    const int seg  = tid * 2 + j;               // 0..255
    const int nrow = seg >> 2;
    const int kseg = (seg & 3) << 3;            // 0,8,16,24 elements
    const bf16_t* g = wT + (size_t)(n0 + nrow) * ldk + kk + kseg;
    const unsigned lds = (unsigned)(buf * BBUFB + (nrow * BROWP + kseg) * 2);
    async_g2l_b128(lds, g, lds_base);
  }
}

// Issue 4 B fragments (batched ds_loads) then 4 WMMAs against one A fragment.
__device__ __forceinline__ void wmma_quad(const bf16_t* bb, v16bf a, v8f* acc, int lane) {
  v16bf b0 = frag_bT_lds(bb + 0 * 16 * BROWP, lane);
  v16bf b1 = frag_bT_lds(bb + 1 * 16 * BROWP, lane);
  v16bf b2 = frag_bT_lds(bb + 2 * 16 * BROWP, lane);
  v16bf b3 = frag_bT_lds(bb + 3 * 16 * BROWP, lane);
  acc[0] = wmma_bf16(a, b0, acc[0]);
  acc[1] = wmma_bf16(a, b1, acc[1]);
  acc[2] = wmma_bf16(a, b2, acc[2]);
  acc[3] = wmma_bf16(a, b3, acc[3]);
}

// Shared async-double-buffered GEMM main loop for the weight GEMMs.
// Computes acc[0..3] (16x64 tile per wave) over K = nk*32, B tile shared by
// the 4 waves of the block via LDS. nk must be even.
__device__ __forceinline__ void gemm_mainloop(const bf16_t* __restrict__ ab, int lda,
                                              const bf16_t* __restrict__ wT, int ldk,
                                              int n0, int nk, int tid, int lane,
                                              bf16_t* bsh, v8f* acc) {
  stage_btile(wT, ldk, n0, 0,  0, tid, bsh);
  stage_btile(wT, ldk, n0, 32, 1, tid, bsh);
  v16bf a0 = frag_a(ab, lda, lane);
  for (int i = 0; i < nk; i += 2) {
    // ---- chunk i (buffer 0); chunk i+1 always already staged
    wait_async_le2();
    __syncthreads();
    v16bf a1 = frag_a(ab + (size_t)(i + 1) * 32, lda, lane);
    wmma_quad(bsh, a0, acc, lane);
    __syncthreads();
    if (i + 2 < nk) stage_btile(wT, ldk, n0, (i + 2) * 32, 0, tid, bsh);
    // ---- chunk i+1 (buffer 1)
    if (i + 2 < nk) wait_async_le2(); else wait_async_0();
    __syncthreads();
    if (i + 2 < nk) a0 = frag_a(ab + (size_t)(i + 2) * 32, lda, lane);
    wmma_quad(bsh + BBUFE, a1, acc, lane);
    __syncthreads();
    if (i + 3 < nk) stage_btile(wT, ldk, n0, (i + 3) * 32, 1, tid, bsh);
  }
}

// ---- fp32 -> bf16 transposing conversion (LDS-tiled) -----------------------
__global__ void __launch_bounds__(256)
cvtT_kernel(const float* __restrict__ src, bf16_t* __restrict__ dst, int K, int N) {
  __shared__ float tile[32][33];
  const size_t soff = (size_t)blockIdx.z * K * N;
  const int n0 = blockIdx.x * 32, k0 = blockIdx.y * 32;
  const int tx = threadIdx.x & 31, ty = threadIdx.x >> 5;   // 32 x 8
  const float* s = src + soff;
  bf16_t*      d = dst + soff;
#pragma unroll
  for (int j = 0; j < 4; ++j)
    tile[ty + 8 * j][tx] = s[(size_t)(k0 + ty + 8 * j) * N + n0 + tx];
  __syncthreads();
#pragma unroll
  for (int j = 0; j < 4; ++j)
    d[(size_t)(n0 + ty + 8 * j) * K + k0 + tx] = (bf16_t)tile[tx][ty + 8 * j];
}

// ---- LayerNorm (optionally fused residual add); emits bf16 -----------------
__device__ __forceinline__ float block_sum256(float v) {
  __shared__ float red[8];
#pragma unroll
  for (int m = 16; m > 0; m >>= 1) v += __shfl_xor(v, m, 32);
  const int wid = threadIdx.x >> 5, lid = threadIdx.x & 31;
  __syncthreads();
  if (lid == 0) red[wid] = v;
  __syncthreads();
  float s = red[0];
#pragma unroll
  for (int i = 1; i < 8; ++i) s += red[i];
  return s;
}

__global__ void __launch_bounds__(256)
ln_kernel(const float* __restrict__ xin, const float* __restrict__ res,
          float* __restrict__ x2out, bf16_t* __restrict__ hout,
          const float* __restrict__ g, const float* __restrict__ beta) {
  const int row = blockIdx.x;
  const float* xr = xin + (size_t)row * D_;
  const float* rr = res ? res + (size_t)row * D_ : nullptr;
  float v[4], s = 0.f;
#pragma unroll
  for (int j = 0; j < 4; ++j) {
    const int c = threadIdx.x + j * 256;
    float t = xr[c];
    if (rr) t += rr[c];
    v[j] = t;
    s += t;
  }
  const float mu = block_sum256(s) * (1.0f / D_);
  float sq = 0.f;
#pragma unroll
  for (int j = 0; j < 4; ++j) { const float d = v[j] - mu; sq += d * d; }
  const float var  = block_sum256(sq) * (1.0f / D_);
  const float rstd = rsqrtf(var + 1e-5f);
  bf16_t* hr  = hout + (size_t)row * D_;
  float*  x2r = x2out ? x2out + (size_t)row * D_ : nullptr;
#pragma unroll
  for (int j = 0; j < 4; ++j) {
    const int c = threadIdx.x + j * 256;
    if (x2r) x2r[c] = v[j];
    hr[c] = (bf16_t)((v[j] - mu) * rstd * g[c] + beta[c]);
  }
}

// ---- QKV projection: 4-wave block computes 64 rows x one head --------------
__global__ void __launch_bounds__(128)
qkv_kernel(const bf16_t* __restrict__ hbf,
           const bf16_t* __restrict__ wqT, const bf16_t* __restrict__ wkT,
           const bf16_t* __restrict__ wvT,
           bf16_t* __restrict__ qo, bf16_t* __restrict__ ko, bf16_t* __restrict__ vT) {
  __shared__ bf16_t bsh[2 * BBUFE];             // double-buffered 64x32 B tile
  const int rt = blockIdx.x, h = blockIdx.y, which = blockIdx.z;
  const int tid = threadIdx.x;
  const int lane = tid & 31, wv = tid >> 5;
  const bf16_t* w = (which == 0) ? wqT : (which == 1) ? wkT : wvT;
  const bf16_t* whT = w + (size_t)h * HS_ * D_;       // [HS][D] transposed
  const int rowA = rt * 64 + wv * 16;
  const bf16_t* ab = hbf + (size_t)rowA * D_;

  v8f acc[4] = {};
  gemm_mainloop(ab, D_, whT, D_, 0, D_ / 32, tid, lane, bsh, acc);

  const int hi = lane >> 4, col = lane & 15;
  const int batch = rowA / T_, t0 = rowA % T_;
  if (which == 2) {
    // vT[b, h, e, t]: rows (t) contiguous per lane -> one 16B store each
    bf16_t* ob = vT + (((size_t)batch * H_ + h) * HS_) * T_;
#pragma unroll
    for (int n = 0; n < 4; ++n) {
      v8bf pk;
#pragma unroll
      for (int r = 0; r < 8; ++r) pk[r] = (bf16_t)acc[n][r];
      *(v8bf*)(ob + (size_t)(n * 16 + col) * T_ + t0 + 8 * hi) = pk;
    }
  } else {
    const float sc = (which == 0) ? 0.125f : 1.0f;    // 1/sqrt(64) into Q
    bf16_t* op = (which == 0) ? qo : ko;
    bf16_t* ob = op + (((size_t)batch * H_ + h) * T_ + t0) * HS_;
#pragma unroll
    for (int n = 0; n < 4; ++n)
#pragma unroll
      for (int r = 0; r < 8; ++r)
        ob[(size_t)(r + 8 * hi) * HS_ + n * 16 + col] = (bf16_t)(acc[n][r] * sc);
  }
}

// ---- Causal flash attention: wave owns one (b, h, 16-row) tile -------------
__global__ void __launch_bounds__(32)
attn_kernel(const bf16_t* __restrict__ q, const bf16_t* __restrict__ kmat,
            const bf16_t* __restrict__ vT, float* __restrict__ att) {
  const int rt = blockIdx.x, h = blockIdx.y, bz = blockIdx.z;
  const int lane = threadIdx.x & 31;
  const int hi = lane >> 4, col = lane & 15;
  __shared__ bf16_t pld[16 * 32];
  const size_t headoff = (((size_t)bz * H_ + h) * T_) * HS_;
  const bf16_t* qb  = q    + headoff;
  const bf16_t* kb  = kmat + headoff;
  const bf16_t* vTb = vT   + headoff;                  // [HS][T] view
  const int row0 = rt * 16;
  const v16bf qa0 = frag_a(qb + (size_t)row0 * HS_,      HS_, lane);  // e 0..31
  const v16bf qa1 = frag_a(qb + (size_t)row0 * HS_ + 32, HS_, lane);  // e 32..63
  v8f oa[4] = {};
  float mrow[8], lrow[8];
#pragma unroll
  for (int r = 0; r < 8; ++r) { mrow[r] = -1e30f; lrow[r] = 0.f; }
  const int nchunk = rt / 2 + 1;
  for (int c = 0; c < nchunk; ++c) {
    const int key0 = c * 32;
    v8f s0 = {}, s1 = {};
    {
      v16bf b0 = frag_b_kT(kb, key0,      0,  lane);
      v16bf b1 = frag_b_kT(kb, key0,      32, lane);
      v16bf b2 = frag_b_kT(kb, key0 + 16, 0,  lane);
      v16bf b3 = frag_b_kT(kb, key0 + 16, 32, lane);
      s0 = wmma_bf16(qa0, b0, s0);
      s0 = wmma_bf16(qa1, b1, s0);
      s1 = wmma_bf16(qa0, b2, s1);
      s1 = wmma_bf16(qa1, b3, s1);
    }
    if (key0 + 15 > row0) {
      const int key = key0 + col;
#pragma unroll
      for (int r = 0; r < 8; ++r)
        if (key > row0 + r + 8 * hi) s0[r] = -1e30f;
    }
    if (key0 + 31 > row0) {
      const int key = key0 + 16 + col;
#pragma unroll
      for (int r = 0; r < 8; ++r)
        if (key > row0 + r + 8 * hi) s1[r] = -1e30f;
    }
#pragma unroll
    for (int r = 0; r < 8; ++r) {
      float mx = fmaxf(s0[r], s1[r]);
#pragma unroll
      for (int m = 8; m > 0; m >>= 1) mx = fmaxf(mx, __shfl_xor(mx, m, 32));
      const float mnew = fmaxf(mrow[r], mx);
      const float corr = __expf(mrow[r] - mnew);
      mrow[r] = mnew;
      const float p0 = __expf(s0[r] - mnew);
      const float p1 = __expf(s1[r] - mnew);
      s0[r] = p0; s1[r] = p1;
      float rs = p0 + p1;
#pragma unroll
      for (int m = 8; m > 0; m >>= 1) rs += __shfl_xor(rs, m, 32);
      lrow[r] = lrow[r] * corr + rs;
#pragma unroll
      for (int n = 0; n < 4; ++n) oa[n][r] *= corr;
    }
    // P (C-layout) -> LDS -> A-layout fragment for P*V
#pragma unroll
    for (int r = 0; r < 8; ++r) {
      const int row = r + 8 * hi;
      pld[row * 32 + col]      = (bf16_t)s0[r];
      pld[row * 32 + 16 + col] = (bf16_t)s1[r];
    }
    __syncthreads();
    v16bf pa;
    {
      const int row = lane & 15;
      const int kbv = hi << 4;
#pragma unroll
      for (int i = 0; i < 16; ++i) pa[i] = pld[row * 32 + kbv + i];
    }
    __syncthreads();
#pragma unroll
    for (int n = 0; n < 4; ++n) {
      v16bf vf = frag_b_vT(vTb, key0, n * 16, lane);
      oa[n] = wmma_bf16(pa, vf, oa[n]);
    }
  }
#pragma unroll
  for (int r = 0; r < 8; ++r) {
    const float inv = 1.0f / lrow[r];
#pragma unroll
    for (int n = 0; n < 4; ++n) oa[n][r] *= inv;
  }
  float* ao = att + ((size_t)bz * T_ + row0) * D_ + h * HS_;
#pragma unroll
  for (int n = 0; n < 4; ++n)
#pragma unroll
    for (int r = 0; r < 8; ++r)
      ao[(size_t)(r + 8 * hi) * D_ + n * 16 + col] = oa[n][r];
}

// ---- MLP GEMM 1: relu(h2 @ w1 + b1) -> bf16 --------------------------------
__global__ void __launch_bounds__(128)
ff1_kernel(const bf16_t* __restrict__ h2, const bf16_t* __restrict__ w1T,
           const float* __restrict__ b1, bf16_t* __restrict__ ff) {
  __shared__ bf16_t bsh[2 * BBUFE];
  const int rt = blockIdx.x, ct = blockIdx.y;
  const int tid = threadIdx.x;
  const int lane = tid & 31, wv = tid >> 5;
  const int rowA = rt * 64 + wv * 16;
  const bf16_t* ab = h2 + (size_t)rowA * D_;

  v8f acc[4] = {};
  gemm_mainloop(ab, D_, w1T, D_, ct * 64, D_ / 32, tid, lane, bsh, acc);

  const int hi = lane >> 4, col = lane & 15;
#pragma unroll
  for (int n = 0; n < 4; ++n) {
    const float bias = b1[ct * 64 + n * 16 + col];
#pragma unroll
    for (int r = 0; r < 8; ++r) {
      const float t = acc[n][r] + bias;
      ff[(size_t)(rowA + r + 8 * hi) * DFF + ct * 64 + n * 16 + col] =
          (bf16_t)fmaxf(t, 0.f);
    }
  }
}

// ---- MLP GEMM 2 + bias + residual -> fp32 output ---------------------------
__global__ void __launch_bounds__(128)
ff2_kernel(const bf16_t* __restrict__ ff, const bf16_t* __restrict__ w2T,
           const float* __restrict__ b2, const float* __restrict__ x2,
           float* __restrict__ out) {
  __shared__ bf16_t bsh[2 * BBUFE];
  const int rt = blockIdx.x, ct = blockIdx.y;
  const int tid = threadIdx.x;
  const int lane = tid & 31, wv = tid >> 5;
  const int rowA = rt * 64 + wv * 16;
  const bf16_t* ab = ff + (size_t)rowA * DFF;

  v8f acc[4] = {};
  gemm_mainloop(ab, DFF, w2T, DFF, ct * 64, DFF / 32, tid, lane, bsh, acc);

  const int hi = lane >> 4, col = lane & 15;
#pragma unroll
  for (int n = 0; n < 4; ++n) {
    const float bias = b2[ct * 64 + n * 16 + col];
#pragma unroll
    for (int r = 0; r < 8; ++r) {
      const size_t idx = (size_t)(rowA + r + 8 * hi) * D_ + ct * 64 + n * 16 + col;
      out[idx] = x2[idx] + acc[n][r] + bias;
    }
  }
}

// ---- host launch -----------------------------------------------------------
extern "C" void kernel_launch(void* const* d_in, const int* in_sizes, int n_in,
                              void* d_out, int out_size, void* d_ws, size_t ws_size,
                              hipStream_t stream) {
  const float* x   = (const float*)d_in[0];
  const float* wq  = (const float*)d_in[1];
  const float* wk  = (const float*)d_in[2];
  const float* wv  = (const float*)d_in[3];
  const float* w1  = (const float*)d_in[4];
  const float* b1  = (const float*)d_in[5];
  const float* w2  = (const float*)d_in[6];
  const float* b2  = (const float*)d_in[7];
  const float* g1  = (const float*)d_in[8];
  const float* be1 = (const float*)d_in[9];
  const float* g2  = (const float*)d_in[10];
  const float* be2 = (const float*)d_in[11];
  float* out = (float*)d_out;

  char* ws = (char*)d_ws;
  size_t off = 0;
  auto alloc = [&](size_t n) -> void* {
    off = (off + 255) & ~(size_t)255;
    void* p = ws + off;
    off += n;
    return p;
  };
  bf16_t* hbf = (bf16_t*)alloc((size_t)NR * D_ * 2);        // ln1 out; reused for ln2
  bf16_t* wqT = (bf16_t*)alloc((size_t)H_ * D_ * HS_ * 2);  // [H][HS][D]
  bf16_t* wkT = (bf16_t*)alloc((size_t)H_ * D_ * HS_ * 2);
  bf16_t* wvT = (bf16_t*)alloc((size_t)H_ * D_ * HS_ * 2);
  bf16_t* w1T = (bf16_t*)alloc((size_t)D_ * DFF * 2);       // [DFF][D]
  bf16_t* w2T = (bf16_t*)alloc((size_t)DFF * D_ * 2);       // [D][DFF]
  bf16_t* qb  = (bf16_t*)alloc((size_t)NR * D_ * 2);        // [B,H,T,HS]
  bf16_t* kb  = (bf16_t*)alloc((size_t)NR * D_ * 2);        // [B,H,T,HS]
  bf16_t* vT  = (bf16_t*)alloc((size_t)NR * D_ * 2);        // [B,H,HS,T]
  float*  att = (float*)alloc((size_t)NR * D_ * 4);
  float*  x2  = (float*)alloc((size_t)NR * D_ * 4);
  bf16_t* ffb = (bf16_t*)alloc((size_t)NR * DFF * 2);

  // Transposing fp32->bf16 weight conversions (per-head slabs for wq/wk/wv)
  cvtT_kernel<<<dim3(HS_ / 32, D_ / 32, H_), 256, 0, stream>>>(wq, wqT, D_, HS_);
  cvtT_kernel<<<dim3(HS_ / 32, D_ / 32, H_), 256, 0, stream>>>(wk, wkT, D_, HS_);
  cvtT_kernel<<<dim3(HS_ / 32, D_ / 32, H_), 256, 0, stream>>>(wv, wvT, D_, HS_);
  cvtT_kernel<<<dim3(DFF / 32, D_ / 32, 1), 256, 0, stream>>>(w1, w1T, D_, DFF);
  cvtT_kernel<<<dim3(D_ / 32, DFF / 32, 1), 256, 0, stream>>>(w2, w2T, DFF, D_);

  // LN1: h = LN(x)
  ln_kernel<<<NR, 256, 0, stream>>>(x, nullptr, nullptr, hbf, g1, be1);
  // QKV (V emitted transposed)
  qkv_kernel<<<dim3(NR / 64, H_, 3), 128, 0, stream>>>(hbf, wqT, wkT, wvT, qb, kb, vT);
  // Attention (writes concat-head fp32)
  attn_kernel<<<dim3(T_ / 16, H_, B_), 32, 0, stream>>>(qb, kb, vT, att);
  // x2 = x + att ; h2 = LN(x2)   (hbf reused as h2)
  ln_kernel<<<NR, 256, 0, stream>>>(x, att, x2, hbf, g2, be2);
  // MLP
  ff1_kernel<<<dim3(NR / 64, DFF / 64), 128, 0, stream>>>(hbf, w1T, b1, ffb);
  ff2_kernel<<<dim3(NR / 64, D_ / 64), 128, 0, stream>>>(ffb, w2T, b2, x2, out);

  (void)in_sizes; (void)n_in; (void)out_size; (void)ws_size;
}
